// CustomCNN_56916906606813
// MI455X (gfx1250) — matrix operand order repaired and checked by reading, not compiled, not run on hardware
//
#include <hip/hip_runtime.h>
#include <hip/hip_bf16.h>

// ---------------------------------------------------------------------------
// Fused LeNet forward for MI455X (gfx1250, wave32).
//
// The reference's "bugged" avg-pool averages over channels AND the 2x2 window
// and broadcasts across channels -> conv2 sees 6 identical channels, conv3
// sees 16 identical channels. So conv2/conv3 run with channel-summed kernels
// (k2s[16,25], k3s[120,25]) on a single-channel map.
//
// conv1 (87% of FLOPs): im2col GEMM on v_wmma_f32_16x16x32_f16
//   A = k1 [16 rows (6 used) x 96 K (75 used)] -> 3 K-steps
//   B = 16 output positions per tile, 49 tiles (49*16 == 784 exactly)
// conv2: one WMMA per 16-position tile (16 out-channels == M=16 exactly).
// conv3/FC1/FC2: tiny VALU dots on LDS-resident activations.
// Image staging: Tensor Data Mover (tensor_load_to_lds) when available,
// tracked with TENSORcnt; vectorized global->LDS fallback otherwise.
// ---------------------------------------------------------------------------

typedef __attribute__((ext_vector_type(16))) _Float16     v16h;
typedef __attribute__((ext_vector_type(8)))  float        v8f;
typedef __attribute__((ext_vector_type(4)))  unsigned int u32x4;
typedef __attribute__((ext_vector_type(8)))  int          i32x8;
typedef __attribute__((ext_vector_type(4)))  int          i32x4;

#define WAVES_PER_WG 2
#define THREADS (WAVES_PER_WG * 32)

#if __has_builtin(__builtin_amdgcn_tensor_load_to_lds) && \
    __has_builtin(__builtin_amdgcn_s_wait_tensorcnt)
#define USE_TDM 1
#else
#define USE_TDM 0
#endif

__device__ __forceinline__ float fast_tanh(float x) {
  // tanh(x) = 1 - 2/(exp(2x)+1); v_exp_f32 + v_rcp_f32 (no IEEE div expansion)
  float e = __expf(2.0f * x);
  return 1.0f - 2.0f * __builtin_amdgcn_rcpf(e + 1.0f);
}

// K index inside a 16-bit 16x32 A-operand for element e (0..15) of the v16h,
// lane-group g (= lane>=16). Per ISA 7.12.2: V0..3 hold K={0..7}+8g,
// V4..7 hold K={16..23}+8g.
__device__ __forceinline__ int a_kidx(int e, int g) {
  return e + ((e >= 8) ? 8 : 0) + g * 8;
}

__global__ __launch_bounds__(THREADS) void lenet_fused(
    const float* __restrict__ x,   // [B,3,32,32]
    const float* __restrict__ k1,  // [6,3,5,5]
    const float* __restrict__ k2,  // [16,6,5,5]
    const float* __restrict__ k3,  // [120,16,5,5]
    const float* __restrict__ W1,  // [120,84]
    const float* __restrict__ b1,  // [84]
    const float* __restrict__ W2,  // [84,10]
    const float* __restrict__ b2,  // [10]
    float* __restrict__ out,       // [B,10]
    int B)
{
  __shared__ float s_img[WAVES_PER_WG][3072];  // 3x32x32 image
  __shared__ float s_t  [WAVES_PER_WG][784];   // conv1 channel-mean map 28x28
  __shared__ float s_m1 [WAVES_PER_WG][196];   // pooled 14x14
  __shared__ float s_c2 [WAVES_PER_WG][100];   // conv2 channel-mean map 10x10
  __shared__ float s_m2 [WAVES_PER_WG][32];    // pooled 5x5
  __shared__ float s_h0 [WAVES_PER_WG][120];   // conv3 output
  __shared__ float s_h1 [WAVES_PER_WG][96];    // fc1 output
  __shared__ float s_k3s[3000];                // channel-folded k3 [120,25]

  const int tid  = threadIdx.x;
  const int w    = tid >> 5;
  const int lane = tid & 31;
  const int m    = lane & 15;   // row / position-in-tile index
  const int g    = lane >> 4;   // lane half-group

  // ---- once per block: fold conv3 weights over the 16 identical channels ---
  for (int i = tid; i < 3000; i += THREADS) {
    int o = i / 25, r = i - o * 25;
    float s = 0.f;
    #pragma unroll
    for (int c = 0; c < 16; ++c) s += k3[o * 400 + c * 25 + r];
    s_k3s[i] = s;
  }

  // ---- once per wave: conv1 weights in WMMA A layout, 3 K-steps of 32 -----
  v16h A1[3];
  #pragma unroll
  for (int st = 0; st < 3; ++st) {
    #pragma unroll
    for (int e = 0; e < 16; ++e) {
      int K = st * 32 + a_kidx(e, g);
      float v = (m < 6 && K < 75) ? k1[m * 75 + K] : 0.f;
      A1[st][e] = (_Float16)v;
    }
  }
  // conv2 channel-folded weights in A layout (K=25 padded to 32, one step)
  v16h A2;
  #pragma unroll
  for (int e = 0; e < 16; ++e) {
    int K = a_kidx(e, g);
    float v = 0.f;
    if (K < 25) {
      #pragma unroll
      for (int c = 0; c < 6; ++c) v += k2[m * 150 + c * 25 + K];
    }
    A2[e] = (_Float16)v;
  }
  __syncthreads();

  const int nwaves = gridDim.x * WAVES_PER_WG;
  const int gwave  = blockIdx.x * WAVES_PER_WG + w;
  const int iters  = (B + nwaves - 1) / nwaves;

  for (int it = 0; it < iters; ++it) {
    int  img    = gwave + it * nwaves;
    bool active = (img < B);
    int  cimg   = active ? img : (B - 1);  // keep barriers uniform

    // ---- stage image into this wave's LDS region (12KB) ------------------
#if USE_TDM
    {
      // Build a 1-D tensor descriptor: 3072 x f32, tile == whole image.
      unsigned int lds_base =
          (unsigned int)(unsigned long long)(&s_img[w][0]);
      unsigned long long ga =
          (unsigned long long)(x + (size_t)cimg * 3072);

      u32x4 g0;
      g0[0] = 1u;  // count=1 (valid user descriptor), no gather, not restore
      g0[1] = (unsigned int)__builtin_amdgcn_readfirstlane((int)lds_base);
      g0[2] = (unsigned int)__builtin_amdgcn_readfirstlane(
                  (int)(unsigned int)(ga & 0xffffffffull));
      g0[3] = (unsigned int)__builtin_amdgcn_readfirstlane(
                  (int)(((unsigned int)(ga >> 32) & 0x01ffffffu) |
                        (2u << 30)));                  // type=2 ("image")

      i32x8 g1;
      g1[0] = (int)(2u << 16);                 // wg_mask=0, data_size=2 (4B)
      g1[1] = (int)((3072u & 0xffffu) << 16);  // tensor_dim0[15:0] @ bits63:48
      g1[2] = (int)((3072u >> 16) | (1u << 16)); // tdim0[31:16], tensor_dim1=1
      g1[3] = (int)(3072u << 16);              // tdim1[31:16]=0, tile_dim0=3072
      g1[4] = 1;                               // tile_dim1=1, tile_dim2=0
      g1[5] = 3072;                            // tensor_dim0_stride[31:0]
      g1[6] = 0;                               // stride hi / dim1_stride lo
      g1[7] = 0;

      i32x4 gz = {0, 0, 0, 0};
#if __clang_major__ >= 23
      i32x8 gz8 = {0, 0, 0, 0, 0, 0, 0, 0};
      __builtin_amdgcn_tensor_load_to_lds(g0, g1, gz, gz, gz8, 0);
#else
      __builtin_amdgcn_tensor_load_to_lds(g0, g1, gz, gz, 0);
#endif
      __builtin_amdgcn_s_wait_tensorcnt(0);
    }
#else
    {
      const float4* xv = (const float4*)x + (size_t)cimg * 768;
      for (int i = lane; i < 768; i += 32)
        ((float4*)s_img[w])[i] = xv[i];
    }
#endif
    __syncthreads();

    // ---- conv1: im2col GEMM via WMMA; tanh + channel mean in epilogue ----
    for (int j = 0; j < 49; ++j) {
      int p  = j * 16 + m;            // output position 0..783
      int oy = p / 28, ox = p - oy * 28;
      v16h Bt[3];
      #pragma unroll
      for (int st = 0; st < 3; ++st) {
        #pragma unroll
        for (int e = 0; e < 16; ++e) {
          int K = st * 32 + a_kidx(e, g);   // B layout: same K order as A per step
          float v = 0.f;
          if (K < 75) {
            int c  = K / 25, r = K - c * 25;
            int dy = r / 5,  dx = r - dy * 5;
            v = s_img[w][c * 1024 + (oy + dy) * 32 + (ox + dx)];
          }
          Bt[st][e] = (_Float16)v;
        }
      }
      v8f acc = {};
      acc = __builtin_amdgcn_wmma_f32_16x16x32_f16(false, A1[0], false, Bt[0],
                                                   (short)0, acc, false, false);
      acc = __builtin_amdgcn_wmma_f32_16x16x32_f16(false, A1[1], false, Bt[1],
                                                   (short)0, acc, false, false);
      acc = __builtin_amdgcn_wmma_f32_16x16x32_f16(false, A1[2], false, Bt[2],
                                                   (short)0, acc, false, false);
      // D layout: lanes 0..15 hold channels 0..7 in acc[0..7] for position p.
      if (lane < 16) {
        float ssum = 0.f;
        #pragma unroll
        for (int r = 0; r < 6; ++r) ssum += fast_tanh(acc[r]);
        s_t[w][p] = ssum * (1.0f / 6.0f);
      }
    }
    __syncthreads();

    // ---- pool1: 2x2 avg of channel-mean map -> 14x14 ---------------------
    for (int idx = lane; idx < 196; idx += 32) {
      int y = idx / 14, xx = idx - y * 14;
      const float* tw = s_t[w];
      s_m1[w][idx] = 0.25f * (tw[(2*y)*28 + 2*xx]     + tw[(2*y)*28 + 2*xx + 1] +
                              tw[(2*y+1)*28 + 2*xx]   + tw[(2*y+1)*28 + 2*xx + 1]);
    }
    __syncthreads();

    // ---- conv2: single-channel conv with k2s, one WMMA per 16 positions --
    for (int j = 0; j < 7; ++j) {
      int p  = j * 16 + m;            // 0..111, valid < 100
      int oy = p / 10, ox = p - oy * 10;
      v16h Bt;
      #pragma unroll
      for (int e = 0; e < 16; ++e) {
        int K = a_kidx(e, g);
        float v = 0.f;
        if (K < 25 && p < 100) {
          int dy = K / 5, dx = K - dy * 5;
          v = s_m1[w][(oy + dy) * 14 + (ox + dx)];
        }
        Bt[e] = (_Float16)v;
      }
      v8f acc = {};
      acc = __builtin_amdgcn_wmma_f32_16x16x32_f16(false, A2, false, Bt,
                                                   (short)0, acc, false, false);
      // rows 0..7 here, rows 8..15 in lane+16: fold halves with a shuffle.
      float s8 = 0.f;
      #pragma unroll
      for (int r = 0; r < 8; ++r) s8 += fast_tanh(acc[r]);
      float full = s8 + __shfl_xor(s8, 16, 32);
      if (lane < 16 && p < 100) s_c2[w][p] = full * (1.0f / 16.0f);
    }
    __syncthreads();

    // ---- pool2: 2x2 avg -> 5x5 ------------------------------------------
    if (lane < 25) {
      int y = lane / 5, xx = lane - y * 5;
      s_m2[w][lane] = 0.25f * (s_c2[w][(2*y)*10 + 2*xx]   + s_c2[w][(2*y)*10 + 2*xx + 1] +
                               s_c2[w][(2*y+1)*10 + 2*xx] + s_c2[w][(2*y+1)*10 + 2*xx + 1]);
    }
    __syncthreads();

    // ---- conv3 (25 -> 120 GEMV with folded k3s) + tanh -------------------
    for (int o = lane; o < 120; o += 32) {
      float a = 0.f;
      #pragma unroll
      for (int K = 0; K < 25; ++K) a += s_k3s[o * 25 + K] * s_m2[w][K];
      s_h0[w][o] = fast_tanh(a);
    }
    __syncthreads();

    // ---- FC1: 120 -> 84, tanh -------------------------------------------
    for (int j = lane; j < 84; j += 32) {
      float a = b1[j];
      for (int i = 0; i < 120; ++i) a += s_h0[w][i] * W1[i * 84 + j];
      s_h1[w][j] = fast_tanh(a);
    }
    __syncthreads();

    // ---- FC2: 84 -> 10 ---------------------------------------------------
    if (lane < 10 && active) {
      float a = b2[lane];
      #pragma unroll
      for (int i = 0; i < 84; ++i) a += s_h1[w][i] * W2[i * 10 + lane];
      out[(size_t)img * 10 + lane] = a;
    }
    __syncthreads();  // s_* regions reused next iteration
  }
}

extern "C" void kernel_launch(void* const* d_in, const int* in_sizes, int n_in,
                              void* d_out, int out_size, void* d_ws, size_t ws_size,
                              hipStream_t stream) {
  const float* x  = (const float*)d_in[0];
  const float* k1 = (const float*)d_in[1];
  const float* k2 = (const float*)d_in[2];
  const float* k3 = (const float*)d_in[3];
  const float* W1 = (const float*)d_in[4];
  const float* b1 = (const float*)d_in[5];
  const float* W2 = (const float*)d_in[6];
  const float* b2 = (const float*)d_in[7];
  float* out = (float*)d_out;

  int B = in_sizes[0] / 3072;            // 3*32*32 floats per image
  if (B <= 0) return;
  // 2 waves per block, ~2 images per wave for B=8192 -> 2048 blocks.
  int blocks = (B + 2 * WAVES_PER_WG - 1) / (2 * WAVES_PER_WG);
  if (blocks < 1) blocks = 1;

  lenet_fused<<<blocks, THREADS, 0, stream>>>(x, k1, k2, k3, W1, b1, W2, b2,
                                              out, B);
}
